// Decoder_4398046511862
// MI455X (gfx1250) — matrix operand order repaired and checked by reading, not compiled, not run on hardware
//
#include <hip/hip_runtime.h>
#include <hip/hip_bf16.h>

// ---------------------------------------------------------------------------
// CDNA5 (gfx1250) seq2seq decoder step.
// GEMMs on v_wmma_f32_16x16x32_bf16 with fp32 = bf16_hi + bf16_lo splitting
// (3 WMMA products, ~1e-5 rel accuracy). Activations pre-split once (reused
// by every N-tile block); weights split during LDS staging (read once).
// Block = 128 threads (4 waves); wave = 2 m-tiles x 4 n-tiles = 24 WMMA per
// 32-wide K-chunk, so each B fragment read from LDS feeds 6 WMMAs.
// ---------------------------------------------------------------------------

typedef __attribute__((ext_vector_type(16))) __bf16          v16bf;
typedef __attribute__((ext_vector_type(16))) unsigned short  v16u;
typedef __attribute__((ext_vector_type(8)))  float           v8f;

union AFrag { uint4 q[2]; v16u u; v16bf b; };

#define B_   128
#define S_   128
#define H_   1024
#define E_   1024   // enc hidden
#define EMB_ 64
#define V_   32000
#define K_   128    // attention proj
#define H4_  4096
#define BH_  (B_ * H_)

static __device__ __forceinline__ unsigned short f32_to_bf16_rne(float x) {
  unsigned u = __float_as_uint(x);
  unsigned r = u + 0x7fffu + ((u >> 16) & 1u);   // round-to-nearest-even
  return (unsigned short)(r >> 16);
}
static __device__ __forceinline__ float bf16us_to_f32(unsigned short h) {
  return __uint_as_float(((unsigned)h) << 16);
}
static __device__ __forceinline__ float sigmoidf_(float x) {
  return 1.f / (1.f + expf(-x));
}
static __device__ __forceinline__ void split2(float x, unsigned short& h,
                                              unsigned short& l) {
  h = f32_to_bf16_rne(x);
  l = f32_to_bf16_rne(x - bf16us_to_f32(h));
}

// ---------------------------------------------------------------------------
// GEMM: C[128, N] (+)= X[128, K] * op(W) + bias0 + bias1
//   WKN==false : W is [N, K] row-major (PyTorch Linear; use W^T)
//   WKN==true  : W is [K, N] row-major
// X supplied pre-split as bf16 hi/lo (ushort). K%32==0, N%64==0.
// Grid: N/64 blocks of 128 threads (4 waves; wave w = batch rows 32w..32w+31).
// ---------------------------------------------------------------------------
template <bool WKN>
__global__ __launch_bounds__(128) void gemm_wmma(
    const unsigned short* __restrict__ Xhi,
    const unsigned short* __restrict__ Xlo, int ldx,
    const float* __restrict__ W, int ldw,
    const float* __restrict__ bias0, const float* __restrict__ bias1,
    float* __restrict__ C, int ldc, int K, int accumulate)
{
  __shared__ unsigned short sBhi[32][64];
  __shared__ unsigned short sBlo[32][64];

  const int tid  = threadIdx.x;
  const int wave = tid >> 5;
  const int lane = tid & 31;
  const int n0 = blockIdx.x << 6;   // 64 columns per block
  const int m0 = wave << 5;         // 32 batch rows per wave (2 m-tiles)

  // A layout (16-bit A 16x32): lane m = lane&15; lanes<16 -> k 0..7 & 16..23,
  // lanes>=16 -> k 8..15 & 24..31.
  const int mrow  = m0 + (lane & 15);
  const int khalf = (lane >> 4) << 3;

  // C/D layout: VGPR r, lanes 0-15 -> M=r, lanes 16-31 -> M=r+8; N = lane&15.
  const int accM  = m0 + ((lane >> 4) << 3);
  const int accNb = lane & 15;

  v8f acc[2][4];
#pragma unroll
  for (int mt = 0; mt < 2; ++mt)
#pragma unroll
    for (int t = 0; t < 4; ++t)
#pragma unroll
      for (int r = 0; r < 8; ++r)
        acc[mt][t][r] =
            accumulate
                ? C[(size_t)(accM + mt * 16 + r) * ldc + (n0 + t * 16 + accNb)]
                : 0.f;

  for (int k0 = 0; k0 < K; k0 += 32) {
    __syncthreads();
    // ---- stage 32k x 64n weight tile as bf16 hi/lo into LDS (B layout [k][n])
    if (WKN) {
      // W[K,N]: thread owns 1k x 16n; lanes 0-3 read 64 consecutive n.
      const int k = tid >> 2;
      const int g = (tid & 3) << 4;   // n offset 0,16,32,48
      const float* wp = W + (size_t)(k0 + k) * ldw + n0 + g;
      float wv[16];
      ((float4*)wv)[0] = *(const float4*)(wp);
      ((float4*)wv)[1] = *(const float4*)(wp + 4);
      ((float4*)wv)[2] = *(const float4*)(wp + 8);
      ((float4*)wv)[3] = *(const float4*)(wp + 12);
      unsigned hp[8], lp[8];
#pragma unroll
      for (int j = 0; j < 8; ++j) {
        unsigned short h0, l0, h1, l1;
        split2(wv[2 * j], h0, l0);
        split2(wv[2 * j + 1], h1, l1);
        hp[j] = (unsigned)h0 | ((unsigned)h1 << 16);
        lp[j] = (unsigned)l0 | ((unsigned)l1 << 16);
      }
      *(uint4*)(&sBhi[k][g])     = make_uint4(hp[0], hp[1], hp[2], hp[3]);
      *(uint4*)(&sBhi[k][g + 8]) = make_uint4(hp[4], hp[5], hp[6], hp[7]);
      *(uint4*)(&sBlo[k][g])     = make_uint4(lp[0], lp[1], lp[2], lp[3]);
      *(uint4*)(&sBlo[k][g + 8]) = make_uint4(lp[4], lp[5], lp[6], lp[7]);
    } else {
      // W[N,K]: thread owns 4k x 4n (4 x b128 row loads, b64 packed stores).
      const int kq = (tid & 7) << 2;  // k offset 0,4,...,28
      const int g  = (tid >> 3) << 2; // n offset 0,4,...,60
      float wv[4][4];
#pragma unroll
      for (int j = 0; j < 4; ++j)
        ((float4*)wv[j])[0] =
            *(const float4*)(W + (size_t)(n0 + g + j) * ldw + k0 + kq);
#pragma unroll
      for (int c = 0; c < 4; ++c) {
        unsigned short h[4], l[4];
#pragma unroll
        for (int j = 0; j < 4; ++j) split2(wv[j][c], h[j], l[j]);
        const uint2 hp = make_uint2((unsigned)h[0] | ((unsigned)h[1] << 16),
                                    (unsigned)h[2] | ((unsigned)h[3] << 16));
        const uint2 lp = make_uint2((unsigned)l[0] | ((unsigned)l[1] << 16),
                                    (unsigned)l[2] | ((unsigned)l[3] << 16));
        *(uint2*)(&sBhi[kq + c][g]) = hp;
        *(uint2*)(&sBlo[kq + c][g]) = lp;
      }
    }
    __syncthreads();

    // ---- A fragments for both m-tiles: pre-split bf16, global_load_b128 only
    AFrag ahi[2], alo[2];
#pragma unroll
    for (int mt = 0; mt < 2; ++mt) {
      const unsigned short* xph =
          Xhi + (size_t)(mrow + mt * 16) * ldx + k0 + khalf;
      const unsigned short* xpl =
          Xlo + (size_t)(mrow + mt * 16) * ldx + k0 + khalf;
      ahi[mt].q[0] = *(const uint4*)(xph);
      ahi[mt].q[1] = *(const uint4*)(xph + 16);
      alo[mt].q[0] = *(const uint4*)(xpl);
      alo[mt].q[1] = *(const uint4*)(xpl + 16);
    }

    // ---- 4 n-tiles x 2 m-tiles x 3 split-products = 24 WMMA
#pragma unroll
    for (int t = 0; t < 4; ++t) {
      AFrag bh, bl;
#pragma unroll
      for (int s = 0; s < 16; ++s) {
        const int kk = khalf + (s & 7) + ((s >> 3) << 4);
        bh.u[s] = sBhi[kk][t * 16 + (lane & 15)];
        bl.u[s] = sBlo[kk][t * 16 + (lane & 15)];
      }
      const v16bf bhi = bh.b;
      const v16bf blo = bl.b;
#pragma unroll
      for (int mt = 0; mt < 2; ++mt) {
        acc[mt][t] = __builtin_amdgcn_wmma_f32_16x16x32_bf16(
            false, ahi[mt].b, false, bhi, (short)0, acc[mt][t], false, false);
        acc[mt][t] = __builtin_amdgcn_wmma_f32_16x16x32_bf16(
            false, alo[mt].b, false, bhi, (short)0, acc[mt][t], false, false);
        acc[mt][t] = __builtin_amdgcn_wmma_f32_16x16x32_bf16(
            false, ahi[mt].b, false, blo, (short)0, acc[mt][t], false, false);
      }
    }
  }

#pragma unroll
  for (int t = 0; t < 4; ++t) {
    const int n = n0 + t * 16 + accNb;
    float badd = 0.f;
    if (bias0) badd += bias0[n];
    if (bias1) badd += bias1[n];
#pragma unroll
    for (int mt = 0; mt < 2; ++mt)
#pragma unroll
      for (int r = 0; r < 8; ++r)
        C[(size_t)(accM + mt * 16 + r) * ldc + n] = acc[mt][t][r] + badd;
  }
}

// ---------------------------------------------------------------------------
// Prepack / elementwise kernels
// ---------------------------------------------------------------------------
__global__ __launch_bounds__(256) void split_kernel(
    const float* __restrict__ src, unsigned short* __restrict__ hi,
    unsigned short* __restrict__ lo, int n)
{
  const int idx = blockIdx.x * 256 + threadIdx.x;
  if (idx >= n) return;
  unsigned short h, l;
  split2(src[idx], h, l);
  hi[idx] = h;
  lo[idx] = l;
}

__global__ __launch_bounds__(256) void embed_concat_split_kernel(
    const int* __restrict__ tokens, const float* __restrict__ E,
    const float* __restrict__ context, unsigned short* __restrict__ hi,
    unsigned short* __restrict__ lo)
{
  const int idx = blockIdx.x * 256 + threadIdx.x;
  if (idx >= B_ * (EMB_ + H_)) return;
  const int b = idx / (EMB_ + H_);
  const int j = idx - b * (EMB_ + H_);
  const float x = (j < EMB_) ? E[(size_t)tokens[b] * EMB_ + j]
                             : context[(size_t)b * H_ + (j - EMB_)];
  unsigned short h, l;
  split2(x, h, l);
  hi[idx] = h;
  lo[idx] = l;
}

__global__ __launch_bounds__(256) void concat2_split_kernel(
    const float* __restrict__ P, const float* __restrict__ Q,
    unsigned short* __restrict__ hi, unsigned short* __restrict__ lo)
{
  const int idx = blockIdx.x * 256 + threadIdx.x;
  if (idx >= B_ * 2048) return;
  const int b = idx >> 11;
  const int j = idx & 2047;
  const float x = (j < 1024) ? P[(b << 10) + j] : Q[(b << 10) + (j - 1024)];
  unsigned short h, l;
  split2(x, h, l);
  hi[idx] = h;
  lo[idx] = l;
}

__global__ __launch_bounds__(256) void lstm_cell_kernel(
    const float* __restrict__ g, const float* __restrict__ cprev,
    float* __restrict__ hout, unsigned short* __restrict__ hhi,
    unsigned short* __restrict__ hlo)
{
  const int idx = blockIdx.x * 256 + threadIdx.x;
  if (idx >= BH_) return;
  const int b = idx >> 10;
  const int j = idx & (H_ - 1);
  const float* gb = g + (size_t)b * H4_;
  const float gi = gb[j];
  const float gf = gb[H_ + j];
  const float gg = gb[2 * H_ + j];
  const float go = gb[3 * H_ + j];
  const float c2 = sigmoidf_(gf) * cprev[idx] + sigmoidf_(gi) * tanhf(gg);
  const float hv = sigmoidf_(go) * tanhf(c2);
  hout[idx] = hv;
  unsigned short h, l;
  split2(hv, h, l);
  hhi[idx] = h;
  hlo[idx] = l;
}

// scores[s,b] = enc[s,b]·v[b] + bae·ad[b]; softmax over s; ctx = sum alphas*enc
__global__ __launch_bounds__(256) void attention_kernel(
    const float* __restrict__ enc, const float* __restrict__ v,
    const float* __restrict__ ad, const float* __restrict__ bae,
    float* __restrict__ ctx)
{
  const int b = blockIdx.x;
  const int tid = threadIdx.x;
  const int wave = tid >> 5, lane = tid & 31;
  __shared__ float red[256];
  __shared__ float sc[S_];
  __shared__ float sval;

  // t = dot(bae, ad[b,:])
  red[tid] = (tid < K_) ? bae[tid] * ad[b * K_ + tid] : 0.f;
  __syncthreads();
  for (int s = 128; s > 0; s >>= 1) {
    if (tid < s) red[tid] += red[tid + s];
    __syncthreads();
  }
  if (tid == 0) sval = red[0];
  __syncthreads();
  const float tdot = sval;

  const float* vp = v + (size_t)b * E_;
  for (int i = 0; i < 16; ++i) {
    const int s = wave * 16 + i;
    const float* ep = enc + ((size_t)s * B_ + b) * E_;
    float p = 0.f;
    for (int e = lane; e < E_; e += 32) p += ep[e] * vp[e];
    for (int off = 16; off; off >>= 1) p += __shfl_xor(p, off, 32);
    if (lane == 0) sc[s] = p + tdot;
  }
  __syncthreads();

  // softmax over S
  red[tid] = (tid < S_) ? sc[tid] : -3.4e38f;
  __syncthreads();
  for (int s = 128; s > 0; s >>= 1) {
    if (tid < s) red[tid] = fmaxf(red[tid], red[tid + s]);
    __syncthreads();
  }
  if (tid == 0) sval = red[0];
  __syncthreads();
  const float mx = sval;
  float ex = 0.f;
  if (tid < S_) { ex = expf(sc[tid] - mx); sc[tid] = ex; }
  red[tid] = (tid < S_) ? ex : 0.f;
  __syncthreads();
  for (int s = 128; s > 0; s >>= 1) {
    if (tid < s) red[tid] += red[tid + s];
    __syncthreads();
  }
  if (tid == 0) sval = red[0];
  __syncthreads();
  const float inv = 1.f / sval;
  if (tid < S_) sc[tid] *= inv;
  __syncthreads();

  for (int e = tid; e < E_; e += 256) {
    float a = 0.f;
    for (int s = 0; s < S_; ++s)
      a += sc[s] * enc[((size_t)s * B_ + b) * E_ + e];
    ctx[(size_t)b * E_ + e] = a;
  }
}

// ---------------------------------------------------------------------------
extern "C" void kernel_launch(void* const* d_in, const int* in_sizes, int n_in,
                              void* d_out, int out_size, void* d_ws, size_t ws_size,
                              hipStream_t stream)
{
  const int*   tokens  = (const int*)  d_in[0];
  const float* hidden  = (const float*)d_in[1];   // [4, B, H]
  const float* cell    = (const float*)d_in[2];   // [4, B, H]
  const float* enc     = (const float*)d_in[3];   // [S, B, E]
  const float* context = (const float*)d_in[4];   // [B, A]
  const float* E       = (const float*)d_in[5];
  const float* W_ih_f  = (const float*)d_in[6];
  const float* W_hh_f  = (const float*)d_in[7];
  const float* b_ih_f  = (const float*)d_in[8];
  const float* b_hh_f  = (const float*)d_in[9];
  const float* W_ih_l0 = (const float*)d_in[10];
  const float* W_hh_l0 = (const float*)d_in[11];
  const float* b_ih_l0 = (const float*)d_in[12];
  const float* b_hh_l0 = (const float*)d_in[13];
  const float* W_ih_r  = (const float*)d_in[14];  // [2, 4096, 1024]
  const float* W_hh_r  = (const float*)d_in[15];
  const float* b_ih_r  = (const float*)d_in[16];
  const float* b_hh_r  = (const float*)d_in[17];
  const float* Wae     = (const float*)d_in[18];  // [128, 1024]
  const float* bae     = (const float*)d_in[19];
  const float* Wad     = (const float*)d_in[20];  // [128, 1024]
  const float* bad     = (const float*)d_in[21];
  const float* Wout    = (const float*)d_in[22];  // [32000, 2048]
  const float* bout    = (const float*)d_in[23];
  float* out = (float*)d_out;

  // ---- workspace carve-up (fp32 first, then bf16-ushort regions)
  float* ws   = (float*)d_ws;
  float* g    = ws;               // 128*4096
  float* o1   = g    + B_ * H4_;  // 128*1024
  float* ad   = o1   + BH_;       // 128*128
  float* vbuf = ad   + B_ * K_;   // 128*1024
  float* ctx  = vbuf + BH_;       // 128*1024
  float* h    = ctx  + BH_;       // 128*1024
  unsigned short* us   = (unsigned short*)(h + BH_);
  unsigned short* x0hi = us;                  // 128*1088
  unsigned short* x0lo = x0hi + B_ * 1088;
  unsigned short* hdhi = x0lo + B_ * 1088;    // 4*128*1024
  unsigned short* hdlo = hdhi + 4 * BH_;
  unsigned short* o1hi = hdlo + 4 * BH_;      // 128*1024
  unsigned short* o1lo = o1hi + BH_;
  unsigned short* adhi = o1lo + BH_;          // 128*128
  unsigned short* adlo = adhi + B_ * K_;
  unsigned short* xchi = adlo + B_ * K_;      // 128*2048
  unsigned short* xclo = xchi + B_ * 2048;
  unsigned short* hhi  = xclo + B_ * 2048;    // 128*1024
  unsigned short* hlo  = hhi  + BH_;

  const dim3 blk(256);
  const dim3 gblk(128);

  // 0) prepack hidden states (all 4 layers) to bf16 hi/lo
  split_kernel<<<dim3(4 * BH_ / 256), blk, 0, stream>>>(hidden, hdhi, hdlo, 4 * BH_);

  // 1) x = E[tokens]; Xcat0 = [x | context] (split)
  embed_concat_split_kernel<<<dim3((B_ * 1088 + 255) / 256), blk, 0, stream>>>(
      tokens, E, context, x0hi, x0lo);

  // 2) LSTM layer 0
  gemm_wmma<false><<<dim3(H4_ / 64), gblk, 0, stream>>>(
      x0hi, x0lo, 1088, W_ih_f, 1088, b_ih_f, b_hh_f, g, H4_, 1088, 0);
  gemm_wmma<false><<<dim3(H4_ / 64), gblk, 0, stream>>>(
      hdhi, hdlo, H_, W_hh_f, H_, nullptr, nullptr, g, H4_, H_, 1);
  lstm_cell_kernel<<<dim3(BH_ / 256), blk, 0, stream>>>(g, cell, o1, o1hi, o1lo);

  // 3) attention: ad = o1@Wad^T + bad ; v = ad@Wae ; scores/softmax/ctx
  gemm_wmma<false><<<dim3(K_ / 64), gblk, 0, stream>>>(
      o1hi, o1lo, H_, Wad, H_, bad, nullptr, ad, K_, H_, 0);
  split_kernel<<<dim3(B_ * K_ / 256), blk, 0, stream>>>(ad, adhi, adlo, B_ * K_);
  gemm_wmma<true><<<dim3(E_ / 64), gblk, 0, stream>>>(
      adhi, adlo, K_, Wae, E_, nullptr, nullptr, vbuf, E_, K_, 0);
  attention_kernel<<<dim3(B_), blk, 0, stream>>>(enc, vbuf, ad, bae, ctx);

  // 4) LSTM layer 1 on [o1 | ctx]
  concat2_split_kernel<<<dim3(B_ * 2048 / 256), blk, 0, stream>>>(o1, ctx, xchi, xclo);
  gemm_wmma<false><<<dim3(H4_ / 64), gblk, 0, stream>>>(
      xchi, xclo, 2048, W_ih_l0, 2048, b_ih_l0, b_hh_l0, g, H4_, 2048, 0);
  gemm_wmma<false><<<dim3(H4_ / 64), gblk, 0, stream>>>(
      hdhi + BH_, hdlo + BH_, H_, W_hh_l0, H_, nullptr, nullptr, g, H4_, H_, 1);
  lstm_cell_kernel<<<dim3(BH_ / 256), blk, 0, stream>>>(g, cell + BH_, h, hhi, hlo);

  // 5) LSTM layers 2 and 3
  for (int i = 0; i < 2; ++i) {
    gemm_wmma<false><<<dim3(H4_ / 64), gblk, 0, stream>>>(
        hhi, hlo, H_, W_ih_r + (size_t)i * H4_ * H_, H_,
        b_ih_r + i * H4_, b_hh_r + i * H4_, g, H4_, H_, 0);
    gemm_wmma<false><<<dim3(H4_ / 64), gblk, 0, stream>>>(
        hdhi + (size_t)(2 + i) * BH_, hdlo + (size_t)(2 + i) * BH_, H_,
        W_hh_r + (size_t)i * H4_ * H_, H_, nullptr, nullptr, g, H4_, H_, 1);
    lstm_cell_kernel<<<dim3(BH_ / 256), blk, 0, stream>>>(
        g, cell + (size_t)(2 + i) * BH_, h, hhi, hlo);
  }

  // 6) out = [h | ctx] @ Wout^T + bout  (dominant GEMM: 262 MB of Wout)
  concat2_split_kernel<<<dim3(B_ * 2048 / 256), blk, 0, stream>>>(h, ctx, xchi, xclo);
  gemm_wmma<false><<<dim3(V_ / 64), gblk, 0, stream>>>(
      xchi, xclo, 2048, Wout, 2048, bout, nullptr, out, V_, 2048, 0);
}
